// LocalAttention_63831803953265
// MI455X (gfx1250) — matrix operand order repaired and checked by reading, not compiled, not run on hardware
//
#include <hip/hip_runtime.h>
#include <hip/hip_fp16.h>

typedef _Float16 f16;
typedef __attribute__((ext_vector_type(16))) _Float16 v16h;
typedef __attribute__((ext_vector_type(8)))  float    v8f;

#define L_Q 2048
#define S_K 2048
#define NH 8
#define NE 64
#define ND 64
#define HALF_CTX 64
#define NEGBIG (-1.0e9f)

// temp * log2(e): fold softmax temperature and base-2 exp into Q scaling
#define QSCALE 0.1803368801111396f

__device__ __forceinline__ float rowReduceMax(float v) {
#pragma unroll
  for (int m = 1; m < 16; m <<= 1) v = fmaxf(v, __shfl_xor(v, m, 32));
  return v;
}
__device__ __forceinline__ float rowReduceSum(float v) {
#pragma unroll
  for (int m = 1; m < 16; m <<= 1) v += __shfl_xor(v, m, 32);
  return v;
}

// A-matrix (16x32 f16) half-index -> K mapping per CDNA5 ISA 7.12.2
__device__ __forceinline__ int aFragK(int hh, int hi) {
  int v = hh >> 1;
  return ((v >= 4) ? 16 : 0) + hi * 8 + 2 * (v & 3) + (hh & 1);
}

__global__ __launch_bounds__(128) void local_attn_wmma(
    const float* __restrict__ Q, const float* __restrict__ K,
    const float* __restrict__ V, const float* __restrict__ AM,
    float* __restrict__ O) {
  const int lane = threadIdx.x & 31;
  const int wave = threadIdx.x >> 5;
  const int tile = blockIdx.x * 4 + wave;          // 1024 tiles
  const int h    = tile & (NH - 1);
  const int l0   = (tile / NH) * 16;
  const int loN  = lane & 15;                      // column / A-row lane
  const int hi   = lane >> 4;                      // lane half

  __shared__ f16 p_lds_all[4][16 * 32];
  f16* p_lds = p_lds_all[wave];

  // ---- Q A-fragments: two 16x32 e-chunks, pre-scaled by temp*log2e ----
  v16h qa[2];
  {
    const float* qrow = Q + ((size_t)(l0 + loN) * NH + h) * NE;
#pragma unroll
    for (int c = 0; c < 2; ++c)
#pragma unroll
      for (int hh = 0; hh < 16; ++hh)
        qa[c][hh] = (f16)(qrow[c * 32 + aFragK(hh, hi)] * QSCALE);
  }

  float mrow[8], srow[8];
  v8f acc[4];
#pragma unroll
  for (int r = 0; r < 8; ++r) { mrow[r] = NEGBIG; srow[r] = 0.0f; }
#pragma unroll
  for (int dt = 0; dt < 4; ++dt) acc[dt] = (v8f)(0.0f);

  for (int t = 0; t < 5; ++t) {
    const int j0 = l0 - HALF_CTX + 32 * t;

    // ---- scores: 16x32 block = two 16x16 WMMA tiles, e accumulated ----
    v8f sc[2];
#pragma unroll
    for (int nt = 0; nt < 2; ++nt) {
      int j  = j0 + nt * 16 + loN;
      int jc = min(max(j, 0), S_K - 1);
      const float* krow = K + ((size_t)jc * NH + h) * NE;
      v8f c = (v8f)(0.0f);
#pragma unroll
      for (int ec = 0; ec < 2; ++ec) {
        v16h kb;                                   // B: K=e per half, N=key lane
#pragma unroll
        for (int hh = 0; hh < 16; ++hh)
          kb[hh] = (f16)krow[ec * 32 + aFragK(hh, hi)];
        c = __builtin_amdgcn_wmma_f32_16x16x32_f16(false, qa[ec], false, kb,
                                                   (short)0, c, false, false);
      }
      sc[nt] = c;
    }

    // ---- band mask + additive mask + online softmax update ----
    float pmax[8];
#pragma unroll
    for (int r = 0; r < 8; ++r) {
      int l = l0 + r + 8 * hi;
      float best = NEGBIG;
#pragma unroll
      for (int nt = 0; nt < 2; ++nt) {
        int j = j0 + nt * 16 + loN;
        bool valid = (j >= 0) & (j < S_K) & (j >= l - HALF_CTX) & (j <= l + HALF_CTX - 1);
        float s = NEGBIG;
        if (valid) s = sc[nt][r] + AM[(size_t)l * S_K + j] * QSCALE;
        sc[nt][r] = s;
        best = fmaxf(best, s);
      }
      pmax[r] = best;
    }

#pragma unroll
    for (int r = 0; r < 8; ++r) {
      float mnew  = fmaxf(mrow[r], rowReduceMax(pmax[r]));
      float scale = __builtin_amdgcn_exp2f(mrow[r] - mnew);
      mrow[r] = mnew;
      float psum = 0.0f;
#pragma unroll
      for (int nt = 0; nt < 2; ++nt) {
        float p = __builtin_amdgcn_exp2f(sc[nt][r] - mnew);
        sc[nt][r] = p;
        psum += p;
      }
      srow[r] = srow[r] * scale + rowReduceSum(psum);
#pragma unroll
      for (int dt = 0; dt < 4; ++dt) acc[dt][r] *= scale;
    }

    // ---- re-stripe P (C/D layout) -> A layout via per-wave LDS ----
#pragma unroll
    for (int r = 0; r < 8; ++r) {
      int row = r + 8 * hi;
      p_lds[row * 32 + loN]      = (f16)sc[0][r];
      p_lds[row * 32 + 16 + loN] = (f16)sc[1][r];
    }
    __builtin_amdgcn_wave_barrier();
    v16h pa;
#pragma unroll
    for (int hh = 0; hh < 16; ++hh)
      pa[hh] = p_lds[loN * 32 + aFragK(hh, hi)];
    __builtin_amdgcn_wave_barrier();

    // ---- acc += P (16x32) x V_block (32x64): 4 WMMAs over d tiles ----
#pragma unroll
    for (int dt = 0; dt < 4; ++dt) {
      v16h vb;                                     // B: K=key offset, N=d
#pragma unroll
      for (int hh = 0; hh < 16; ++hh) {
        int j  = j0 + aFragK(hh, hi);
        int jc = min(max(j, 0), S_K - 1);
        vb[hh] = (f16)V[((size_t)jc * NH + h) * ND + dt * 16 + loN];
      }
      acc[dt] = __builtin_amdgcn_wmma_f32_16x16x32_f16(false, pa, false, vb,
                                                       (short)0, acc[dt], false, false);
    }
  }

  // ---- normalize and store ----
#pragma unroll
  for (int r = 0; r < 8; ++r) {
    int l = l0 + r + 8 * hi;
    float inv = 1.0f / srow[r];
    float* orow = O + ((size_t)l * NH + h) * ND;
#pragma unroll
    for (int dt = 0; dt < 4; ++dt)
      orow[dt * 16 + loN] = acc[dt][r] * inv;
  }
}

extern "C" void kernel_launch(void* const* d_in, const int* in_sizes, int n_in,
                              void* d_out, int out_size, void* d_ws, size_t ws_size,
                              hipStream_t stream) {
  (void)in_sizes; (void)n_in; (void)d_ws; (void)ws_size; (void)out_size;
  const float* Q  = (const float*)d_in[0];
  const float* K  = (const float*)d_in[1];
  const float* V  = (const float*)d_in[2];
  const float* AM = (const float*)d_in[3];
  float* O = (float*)d_out;
  dim3 grid((NH * (L_Q / 16)) / 4);   // 256 blocks x 4 waves = 1024 tiles
  dim3 block(128);
  hipLaunchKernelGGL(local_attn_wmma, grid, block, 0, stream, Q, K, V, AM, O);
}
